// FullGCNModel_49976239456895
// MI455X (gfx1250) — compile-verified, matrix operand order checked by reading, not run on hardware
//
// CDNA5 (gfx1250) heterogeneous GCN: l2norm -> proj -> 2x GraphConv(mean) ->
// LN+ReLU -> l2norm -> 3 MLP heads (relu, sigmoid).
// Matrix math: v_wmma_f32_16x16x32_f16 (f16 operands, f32 accum), double-
// buffered LDS pipeline with branch-free (clamped) tile loads.
// Scatter-mean uses global f32 atomics (node buffers are L2-resident, 192MB L2).
#include <hip/hip_runtime.h>
#include <hip/hip_fp16.h>
#include <math.h>

typedef __attribute__((ext_vector_type(16))) _Float16 v16h;
typedef __attribute__((ext_vector_type(8)))  _Float16 v8h;
typedef __attribute__((ext_vector_type(8)))  float    v8f;

#define NC_  50000
#define NS_  20000
#define NG_  1000
#define DC_  128
#define DSK_ 128
#define DG_  64
#define CH_  256
#define NCAT_ 1000
#define NSKU_ 2000
#define EBUY_ 300000
#define EBEL_ 40000

__device__ __forceinline__ float wave_sum(float v) {
#pragma unroll
  for (int off = 16; off > 0; off >>= 1) v += __shfl_xor(v, off, 32);
  return v;
}

// ---------------- elementwise / reduction kernels ----------------

__global__ void k_zero(float* __restrict__ p, size_t n) {
  size_t i = (size_t)blockIdx.x * blockDim.x + threadIdx.x;
  size_t st = (size_t)gridDim.x * blockDim.x;
  for (; i < n; i += st) p[i] = 0.0f;
}

// f32 W[K][N] -> f16 Wt[N][K]
__global__ void k_cvt_t(const float* __restrict__ W, _Float16* __restrict__ Wt,
                        int K, int N) {
  size_t total = (size_t)K * N;
  size_t i = (size_t)blockIdx.x * blockDim.x + threadIdx.x;
  size_t st = (size_t)gridDim.x * blockDim.x;
  for (; i < total; i += st) {
    int k = (int)(i / N);
    int n = (int)(i % N);
    Wt[(size_t)n * K + k] = (_Float16)W[i];
  }
}

// per-row L2 normalize; blockDim.x == d (64/128/256). Optional f32 mirror.
__global__ void k_rownorm(const float* __restrict__ x, _Float16* __restrict__ y16,
                          float* __restrict__ y32, int d) {
  __shared__ float red[8];
  const int row = blockIdx.x;
  const int tid = threadIdx.x;
  const float v = x[(size_t)row * d + tid];
  const float s2 = wave_sum(v * v);
  const int lane = tid & 31, wv = tid >> 5, nw = blockDim.x >> 5;
  if (lane == 0) red[wv] = s2;
  __syncthreads();
  float tot = 0.0f;
  for (int i = 0; i < nw; ++i) tot += red[i];
  float nrm = sqrtf(tot);
  nrm = nrm > 1e-12f ? nrm : 1e-12f;
  const float y = v / nrm;
  if (y32) y32[(size_t)row * d + tid] = y;
  y16[(size_t)row * d + tid] = (_Float16)y;
}

// layernorm(eps=1e-5) + relu; blockDim.x == 256 == CH. Writes f32 + f16.
__global__ void k_ln_relu(const float* __restrict__ x, const float* __restrict__ g,
                          const float* __restrict__ b, float* __restrict__ y32,
                          _Float16* __restrict__ y16) {
  __shared__ float red[16];
  const int row = blockIdx.x;
  const int tid = threadIdx.x;
  const size_t idx = (size_t)row * CH_ + tid;
  const float v = x[idx];
  const float s = wave_sum(v);
  const float s2 = wave_sum(v * v);
  const int lane = tid & 31, wv = tid >> 5;
  if (lane == 0) { red[wv] = s; red[8 + wv] = s2; }
  __syncthreads();
  float sum = 0.0f, sumsq = 0.0f;
#pragma unroll
  for (int i = 0; i < 8; ++i) { sum += red[i]; sumsq += red[8 + i]; }
  const float mu = sum * (1.0f / CH_);
  float var = sumsq * (1.0f / CH_) - mu * mu;
  var = var > 0.0f ? var : 0.0f;
  const float inv = rsqrtf(var + 1e-5f);
  float y = (v - mu) * inv * g[tid] + b[tid];
  y = y > 0.0f ? y : 0.0f;
  y32[idx] = y;
  y16[idx] = (_Float16)y;
}

// edge scatter-add: ei = [src[E], dst[E]]. blockDim == 256 == CH.
__global__ void k_scatter(const float* __restrict__ xsrc, const int* __restrict__ ei,
                          int E, float* __restrict__ sums, float* __restrict__ cnt) {
  const int tid = threadIdx.x;
  const int e0 = blockIdx.x * 4;
#pragma unroll
  for (int i = 0; i < 4; ++i) {
    const int e = e0 + i;
    if (e >= E) break;
    const int s = ei[e];
    const int d = ei[E + e];
    atomicAdd(&sums[(size_t)d * CH_ + tid], xsrc[(size_t)s * CH_ + tid]);
    if (tid == 0) atomicAdd(&cnt[d], 1.0f);
  }
}

// mean = sums / max(cnt,1), emitted directly as f16 GEMM operand
__global__ void k_mean(const float* __restrict__ sums, const float* __restrict__ cnt,
                       _Float16* __restrict__ out, size_t total) {
  size_t i = (size_t)blockIdx.x * blockDim.x + threadIdx.x;
  size_t st = (size_t)gridDim.x * blockDim.x;
  for (; i < total; i += st) {
    float c = cnt[i >> 8];
    c = c < 1.0f ? 1.0f : c;
    out[i] = (_Float16)(sums[i] / c);
  }
}

// ---------------- WMMA GEMM ----------------
// C[M][N] (+)= A16[M][K] * Bt16[N][K]^T + bias, optional relu/sigmoid epilogue,
// optional f16 mirror. 128x128x32 block tile, 8 waves, each wave owns 32x64
// (2x4 tiles of 16x16) of v_wmma_f32_16x16x32_f16 accumulators.
// Double-buffered LDS; out-of-range tile rows are CLAMPED (their products land
// in never-stored accumulator rows/cols), so the fill path is branch-free.
__device__ __forceinline__ v16h ld_frag(const _Float16* rowbase, int khalf) {
  const v8h lo = *(const v8h*)(rowbase + khalf * 8);
  const v8h hi = *(const v8h*)(rowbase + khalf * 8 + 16);
  v16h f;
#pragma unroll
  for (int i = 0; i < 8; ++i) { f[i] = lo[i]; f[i + 8] = hi[i]; }
  return f;
}

__global__ __launch_bounds__(256)
void k_gemm(const _Float16* __restrict__ A, const _Float16* __restrict__ Bt,
            const float* __restrict__ bias, float* __restrict__ C,
            _Float16* __restrict__ Ch, int M, int N, int K,
            int accumulate, int epilogue /*0 none,1 relu,2 sigmoid*/) {
  constexpr int BM = 128, BN = 128, BK = 32, LDA = BK + 8; // 80B row, 16B-aligned
  __shared__ __align__(16) _Float16 sA[2][BM * LDA];
  __shared__ __align__(16) _Float16 sB[2][BN * LDA];
  const int tid  = threadIdx.x;
  const int lane = tid & 31;
  const int wv   = tid >> 5;
  const int wr   = wv >> 1;   // 0..3 -> row quadrant
  const int wc   = wv & 1;    // 0..1 -> col half
  const int m0   = blockIdx.x * BM;
  const int n0   = blockIdx.y * BN;

  // branch-free tile staging: each thread owns two 8-f16 chunks per operand
  const int rloc = tid >> 2;             // 0..63 (+64 for second chunk)
  const int kk   = (tid & 3) << 3;       // 0/8/16/24
  int ra0 = m0 + rloc;        if (ra0 >= M) ra0 = M - 1;
  int ra1 = m0 + 64 + rloc;   if (ra1 >= M) ra1 = M - 1;
  int rb0 = n0 + rloc;        if (rb0 >= N) rb0 = N - 1;
  int rb1 = n0 + 64 + rloc;   if (rb1 >= N) rb1 = N - 1;
  const _Float16* pa0 = A + (size_t)ra0 * K + kk;
  const _Float16* pa1 = A + (size_t)ra1 * K + kk;
  const _Float16* pb0 = Bt + (size_t)rb0 * K + kk;
  const _Float16* pb1 = Bt + (size_t)rb1 * K + kk;

  v8f acc[2][4];
#pragma unroll
  for (int i = 0; i < 2; ++i)
#pragma unroll
    for (int j = 0; j < 4; ++j)
#pragma unroll
      for (int r = 0; r < 8; ++r) acc[i][j][r] = 0.0f;

  v8h a0, a1, b0, b1;
  // prologue: tile 0 -> regs -> LDS buffer 0
  a0 = *(const v8h*)(pa0); a1 = *(const v8h*)(pa1);
  b0 = *(const v8h*)(pb0); b1 = *(const v8h*)(pb1);
  *(v8h*)(sA[0] + rloc * LDA + kk)        = a0;
  *(v8h*)(sA[0] + (64 + rloc) * LDA + kk) = a1;
  *(v8h*)(sB[0] + rloc * LDA + kk)        = b0;
  *(v8h*)(sB[0] + (64 + rloc) * LDA + kk) = b1;

  const int nt = K / BK;
  const int khalf = lane >> 4;
  const int lr = lane & 15;
  for (int t = 0; t < nt; ++t) {
    __syncthreads();
    const int buf = t & 1;
    // prefetch next K-tile while WMMAs run on the current one
    if (t + 1 < nt) {
      const int ko = (t + 1) * BK;
      a0 = *(const v8h*)(pa0 + ko); a1 = *(const v8h*)(pa1 + ko);
      b0 = *(const v8h*)(pb0 + ko); b1 = *(const v8h*)(pb1 + ko);
    }
    v16h af[2], bf[4];
#pragma unroll
    for (int i = 0; i < 2; ++i)
      af[i] = ld_frag(sA[buf] + (wr * 32 + i * 16 + lr) * LDA, khalf);
#pragma unroll
    for (int j = 0; j < 4; ++j)
      bf[j] = ld_frag(sB[buf] + (wc * 64 + j * 16 + lr) * LDA, khalf);
#pragma unroll
    for (int i = 0; i < 2; ++i)
#pragma unroll
      for (int j = 0; j < 4; ++j)
        acc[i][j] = __builtin_amdgcn_wmma_f32_16x16x32_f16(
            false, af[i], false, bf[j], (short)0, acc[i][j], false, false);
    if (t + 1 < nt) {
      const int nb = buf ^ 1;
      *(v8h*)(sA[nb] + rloc * LDA + kk)        = a0;
      *(v8h*)(sA[nb] + (64 + rloc) * LDA + kk) = a1;
      *(v8h*)(sB[nb] + rloc * LDA + kk)        = b0;
      *(v8h*)(sB[nb] + (64 + rloc) * LDA + kk) = b1;
    }
  }

  // epilogue: lane<16 -> col, VGPR r -> row (+8 for lane>=16)
  const int col = lane & 15;
  const int rh  = lane >> 4;
#pragma unroll
  for (int j = 0; j < 4; ++j) {
    const int n = n0 + wc * 64 + j * 16 + col;
    if (n >= N) continue;
    const float bv = bias ? bias[n] : 0.0f;
#pragma unroll
    for (int i = 0; i < 2; ++i) {
      const int mb = m0 + wr * 32 + i * 16 + rh * 8;
#pragma unroll
      for (int r = 0; r < 8; ++r) {
        const int m = mb + r;
        if (m >= M) continue;
        const size_t idx = (size_t)m * N + n;
        float v = acc[i][j][r] + bv;
        if (accumulate) v += C[idx];
        if (epilogue == 1) v = v > 0.0f ? v : 0.0f;
        else if (epilogue == 2) v = 1.0f / (1.0f + __expf(-v));
        if (C)  C[idx] = v;
        if (Ch) Ch[idx] = (_Float16)v;
      }
    }
  }
}

// ---------------- host orchestration ----------------

static inline int grid1d(size_t n, int tpb) {
  size_t b = (n + tpb - 1) / tpb;
  if (b > 1048576) b = 1048576;
  if (b < 1) b = 1;
  return (int)b;
}

extern "C" void kernel_launch(void* const* d_in, const int* in_sizes, int n_in,
                              void* d_out, int out_size, void* d_ws, size_t ws_size,
                              hipStream_t stream) {
  (void)in_sizes; (void)n_in; (void)out_size; (void)ws_size;
  const float* x_client = (const float*)d_in[0];
  const float* x_sku    = (const float*)d_in[1];
  const float* x_cat    = (const float*)d_in[2];
  const int* ei_buys        = (const int*)d_in[3];
  const int* ei_rev_buys    = (const int*)d_in[4];
  const int* ei_belongs     = (const int*)d_in[5];
  const int* ei_rev_belongs = (const int*)d_in[6];
  const float* Win_c = (const float*)d_in[7];  const float* bin_c = (const float*)d_in[8];
  const float* Win_s = (const float*)d_in[9];  const float* bin_s = (const float*)d_in[10];
  const float* Win_g = (const float*)d_in[11]; const float* bin_g = (const float*)d_in[12];
  const float* Wrel  = (const float*)d_in[13];
  const float* brel  = (const float*)d_in[14];
  const float* Wroot = (const float*)d_in[15];
  const float* nrm_g = (const float*)d_in[16];
  const float* nrm_b = (const float*)d_in[17];
  const float* Wch1 = (const float*)d_in[18]; const float* bch1 = (const float*)d_in[19];
  const float* Wch2 = (const float*)d_in[20]; const float* bch2 = (const float*)d_in[21];
  const float* Wcat1 = (const float*)d_in[22]; const float* bcat1 = (const float*)d_in[23];
  const float* Wcat2 = (const float*)d_in[24]; const float* bcat2 = (const float*)d_in[25];
  const float* Wsku1 = (const float*)d_in[26]; const float* bsku1 = (const float*)d_in[27];
  const float* Wsku2 = (const float*)d_in[28]; const float* bsku2 = (const float*)d_in[29];
  float* outp = (float*)d_out;

  // workspace carve-out (~334 MB)
  char* base = (char*)d_ws;
  size_t cur = 0;
  auto alloc = [&](size_t bytes) -> char* {
    cur = (cur + 255) & ~(size_t)255;
    char* p = base + cur;
    cur += bytes;
    return p;
  };
  _Float16* xcn16 = (_Float16*)alloc((size_t)NC_ * DC_ * 2);
  _Float16* xsn16 = (_Float16*)alloc((size_t)NS_ * DSK_ * 2);
  _Float16* xgn16 = (_Float16*)alloc((size_t)NG_ * DG_ * 2);
  float* xc32 = (float*)alloc((size_t)NC_ * CH_ * 4);
  float* xs32 = (float*)alloc((size_t)NS_ * CH_ * 4);
  float* xg32 = (float*)alloc((size_t)NG_ * CH_ * 4);
  _Float16* xc16 = (_Float16*)alloc((size_t)NC_ * CH_ * 2);
  _Float16* xs16 = (_Float16*)alloc((size_t)NS_ * CH_ * 2);
  _Float16* xg16 = (_Float16*)alloc((size_t)NG_ * CH_ * 2);
  float* nc32 = (float*)alloc((size_t)NC_ * CH_ * 4);
  float* ns32 = (float*)alloc((size_t)NS_ * CH_ * 4);
  float* ng32 = (float*)alloc((size_t)NG_ * CH_ * 4);
  float* sum_c = (float*)alloc((size_t)NC_ * CH_ * 4);
  float* sum_s = (float*)alloc((size_t)NS_ * CH_ * 4);
  float* sum_g = (float*)alloc((size_t)NG_ * CH_ * 4);
  float* cnt_c = (float*)alloc((size_t)NC_ * 4);
  float* cnt_s = (float*)alloc((size_t)NS_ * 4);
  float* cnt_g = (float*)alloc((size_t)NG_ * 4);
  _Float16* mean_c16 = (_Float16*)alloc((size_t)NC_ * CH_ * 2);
  _Float16* mean_s16 = (_Float16*)alloc((size_t)NS_ * CH_ * 2);
  _Float16* mean_g16 = (_Float16*)alloc((size_t)NG_ * CH_ * 2);
  _Float16* user16   = (_Float16*)alloc((size_t)NC_ * CH_ * 2);
  _Float16* hid16    = (_Float16*)alloc((size_t)NC_ * 128 * 2);
  _Float16* wt_in_c = (_Float16*)alloc((size_t)DC_ * CH_ * 2);
  _Float16* wt_in_s = (_Float16*)alloc((size_t)DSK_ * CH_ * 2);
  _Float16* wt_in_g = (_Float16*)alloc((size_t)DG_ * CH_ * 2);
  _Float16* wt_rel  = (_Float16*)alloc((size_t)8 * CH_ * CH_ * 2);
  _Float16* wt_root = (_Float16*)alloc((size_t)8 * CH_ * CH_ * 2);
  _Float16* wt_ch1  = (_Float16*)alloc((size_t)CH_ * 128 * 2);
  _Float16* wt_ch2  = (_Float16*)alloc((size_t)128 * 1 * 2);
  _Float16* wt_cat1 = (_Float16*)alloc((size_t)CH_ * 128 * 2);
  _Float16* wt_cat2 = (_Float16*)alloc((size_t)128 * NCAT_ * 2);
  _Float16* wt_sku1 = (_Float16*)alloc((size_t)CH_ * 128 * 2);
  _Float16* wt_sku2 = (_Float16*)alloc((size_t)128 * NSKU_ * 2);

  auto cvt = [&](const float* W, _Float16* Wt, int K, int N) {
    k_cvt_t<<<grid1d((size_t)K * N, 256), 256, 0, stream>>>(W, Wt, K, N);
  };
  auto gemm = [&](const _Float16* A, const _Float16* Bt, const float* bias,
                  float* C, _Float16* Ch, int M, int N, int K, int acc, int ep) {
    dim3 g((M + 127) / 128, (N + 127) / 128);
    k_gemm<<<g, 256, 0, stream>>>(A, Bt, bias, C, Ch, M, N, K, acc, ep);
  };
  auto aggregate = [&](const float* xsrc, const int* ei, int E, float* sums,
                       float* cnt, _Float16* mean16, int ndst) {
    k_zero<<<grid1d((size_t)ndst * CH_, 256), 256, 0, stream>>>(sums, (size_t)ndst * CH_);
    k_zero<<<grid1d((size_t)ndst, 256), 256, 0, stream>>>(cnt, (size_t)ndst);
    k_scatter<<<(E + 3) / 4, 256, 0, stream>>>(xsrc, ei, E, sums, cnt);
    k_mean<<<grid1d((size_t)ndst * CH_, 256), 256, 0, stream>>>(sums, cnt, mean16,
                                                                (size_t)ndst * CH_);
  };

  // 1) weight conversion / transposition (f32 [K][N] -> f16 [N][K])
  cvt(Win_c, wt_in_c, DC_, CH_);
  cvt(Win_s, wt_in_s, DSK_, CH_);
  cvt(Win_g, wt_in_g, DG_, CH_);
  for (int i = 0; i < 8; ++i) {
    cvt(Wrel  + (size_t)i * CH_ * CH_, wt_rel  + (size_t)i * CH_ * CH_, CH_, CH_);
    cvt(Wroot + (size_t)i * CH_ * CH_, wt_root + (size_t)i * CH_ * CH_, CH_, CH_);
  }
  cvt(Wch1, wt_ch1, CH_, 128);   cvt(Wch2, wt_ch2, 128, 1);
  cvt(Wcat1, wt_cat1, CH_, 128); cvt(Wcat2, wt_cat2, 128, NCAT_);
  cvt(Wsku1, wt_sku1, CH_, 128); cvt(Wsku2, wt_sku2, 128, NSKU_);

  // 2) l2-normalize raw features into f16 GEMM operands
  k_rownorm<<<NC_, DC_, 0, stream>>>(x_client, xcn16, nullptr, DC_);
  k_rownorm<<<NS_, DSK_, 0, stream>>>(x_sku, xsn16, nullptr, DSK_);
  k_rownorm<<<NG_, DG_, 0, stream>>>(x_cat, xgn16, nullptr, DG_);

  // 3) input projections (f32 + f16 outputs)
  gemm(xcn16, wt_in_c, bin_c, xc32, xc16, NC_, CH_, DC_, 0, 0);
  gemm(xsn16, wt_in_s, bin_s, xs32, xs16, NS_, CH_, DSK_, 0, 0);
  gemm(xgn16, wt_in_g, bin_g, xg32, xg16, NG_, CH_, DG_, 0, 0);

  // 4) two GCN layers
  for (int l = 0; l < 2; ++l) {
    auto rel  = [&](int e) { return wt_rel  + (size_t)(l * 4 + e) * CH_ * CH_; };
    auto root = [&](int e) { return wt_root + (size_t)(l * 4 + e) * CH_ * CH_; };
    auto bre  = [&](int e) { return brel + (size_t)(l * 4 + e) * CH_; };
    // nc: rev_buys (sku -> client)
    aggregate(xs32, ei_rev_buys, EBUY_, sum_c, cnt_c, mean_c16, NC_);
    gemm(mean_c16, rel(1), bre(1), nc32, nullptr, NC_, CH_, CH_, 0, 0);
    gemm(xc16,     root(1), nullptr, nc32, nullptr, NC_, CH_, CH_, 1, 0);
    // ns: buys (client -> sku) + rev_belongs (cat -> sku)
    aggregate(xc32, ei_buys, EBUY_, sum_s, cnt_s, mean_s16, NS_);
    gemm(mean_s16, rel(0), bre(0), ns32, nullptr, NS_, CH_, CH_, 0, 0);
    gemm(xs16,     root(0), nullptr, ns32, nullptr, NS_, CH_, CH_, 1, 0);
    aggregate(xg32, ei_rev_belongs, EBEL_, sum_s, cnt_s, mean_s16, NS_);
    gemm(mean_s16, rel(3), bre(3), ns32, nullptr, NS_, CH_, CH_, 1, 0);
    gemm(xs16,     root(3), nullptr, ns32, nullptr, NS_, CH_, CH_, 1, 0);
    // ng: belongs (sku -> cat)
    aggregate(xs32, ei_belongs, EBEL_, sum_g, cnt_g, mean_g16, NG_);
    gemm(mean_g16, rel(2), bre(2), ng32, nullptr, NG_, CH_, CH_, 0, 0);
    gemm(xg16,     root(2), nullptr, ng32, nullptr, NG_, CH_, CH_, 1, 0);
    // layernorm + relu (writes next-layer f32 and f16 operands)
    k_ln_relu<<<NC_, CH_, 0, stream>>>(nc32, nrm_g + (size_t)(l * 3 + 0) * CH_,
                                       nrm_b + (size_t)(l * 3 + 0) * CH_, xc32, xc16);
    k_ln_relu<<<NS_, CH_, 0, stream>>>(ns32, nrm_g + (size_t)(l * 3 + 1) * CH_,
                                       nrm_b + (size_t)(l * 3 + 1) * CH_, xs32, xs16);
    k_ln_relu<<<NG_, CH_, 0, stream>>>(ng32, nrm_g + (size_t)(l * 3 + 2) * CH_,
                                       nrm_b + (size_t)(l * 3 + 2) * CH_, xg32, xg16);
  }

  // 5) user embedding (f32 straight into d_out, f16 for head GEMMs)
  const size_t churn_off = 0;
  const size_t cat_off   = (size_t)NC_;
  const size_t sku_off   = cat_off + (size_t)NC_ * NCAT_;
  const size_t user_off  = sku_off + (size_t)NC_ * NSKU_;
  k_rownorm<<<NC_, CH_, 0, stream>>>(xc32, user16, outp + user_off, CH_);

  // 6) heads: hidden = relu(user @ W1 + b1) in f16, out = sigmoid(hidden @ W2 + b2)
  gemm(user16, wt_ch1, bch1, nullptr, hid16, NC_, 128, CH_, 0, 1);
  gemm(hid16, wt_ch2, bch2, outp + churn_off, nullptr, NC_, 1, 128, 0, 2);
  gemm(user16, wt_cat1, bcat1, nullptr, hid16, NC_, 128, CH_, 0, 1);
  gemm(hid16, wt_cat2, bcat2, outp + cat_off, nullptr, NC_, NCAT_, 128, 0, 2);
  gemm(user16, wt_sku1, bsku1, nullptr, hid16, NC_, 128, CH_, 0, 1);
  gemm(hid16, wt_sku2, bsku2, outp + sku_off, nullptr, NC_, NSKU_, 128, 0, 2);
}
